// TransformerBlock_5334349381975
// MI455X (gfx1250) — compile-verified, matrix operand order checked by reading, not run on hardware
//
#include <hip/hip_runtime.h>
#include <hip/hip_bf16.h>
#include <math.h>

// ---------------------------------------------------------------------------
// Types for CDNA5 WMMA (wave32): v_wmma_f32_16x16x32_bf16
// ---------------------------------------------------------------------------
typedef __bf16  v16bf __attribute__((ext_vector_type(16)));
typedef __bf16  v8bf  __attribute__((ext_vector_type(8)));
typedef float   v8f   __attribute__((ext_vector_type(8)));
typedef int     v4i   __attribute__((ext_vector_type(4)));

#define WMMA_BF16(a, b, c) \
    __builtin_amdgcn_wmma_f32_16x16x32_bf16(false, (a), false, (b), (short)0, (c), false, false)

// CDNA5 async global->LDS DMA path (guarded; falls back to uint4 copies).
// Probe round 2 revealed the builtin signature:
//   param0: AS(1) v4i*  (global source), param1: LDS v4i* (dest), then (offset, cpol).
#if __has_builtin(__builtin_amdgcn_global_load_async_to_lds_b128)
#define HAVE_ASYNC_LDS 1
typedef __attribute__((address_space(1))) v4i* as1_v4i;
typedef __attribute__((address_space(3))) v4i* as3_v4i;
__device__ inline void async_copy_b128(const __bf16* g, __bf16* l) {
    __builtin_amdgcn_global_load_async_to_lds_b128(
        (as1_v4i)(unsigned long long)g,   // AS1 address == flat address; drops const
        (as3_v4i)l,                       // generic -> LDS addrspacecast
        0, 0);
}
#define ASYNC_WAIT() asm volatile("s_wait_asynccnt 0x0" ::: "memory")
#else
#define ASYNC_WAIT()
#endif

// Problem sizes (fixed by the reference)
#define BN_  4
#define T_   2048
#define D_   1024
#define H_   16
#define HD_  64
#define D4_  4096
#define M_   (BN_ * T_)   // 8192 rows of activations

// ---------------------------------------------------------------------------
// Fragment loader: both A and B operands are stored so that each lane's
// 16 bf16 values are two contiguous 16-byte runs (ISA 7.12.2 K-octet layout).
// ---------------------------------------------------------------------------
__device__ inline v16bf frag_row16(const __bf16* row, int lane) {
    const int ks = (lane >> 4) & 1;
    const v8bf lo = *(const v8bf*)(row + ks * 8);
    const v8bf hi = *(const v8bf*)(row + 16 + ks * 8);
    return __builtin_shufflevector(lo, hi, 0, 1, 2, 3, 4, 5, 6, 7,
                                           8, 9, 10, 11, 12, 13, 14, 15);
}
__device__ inline v8f v8f_zero() {
    v8f c;
#pragma unroll
    for (int i = 0; i < 8; ++i) c[i] = 0.f;
    return c;
}

// ---------------------------------------------------------------------------
// Weight prep: fp32 [K][N] -> bf16 transposed [N][K] (tiled via LDS)
// ---------------------------------------------------------------------------
__global__ __launch_bounds__(256) void transpose_f32_to_bf16_kernel(
        const float* __restrict__ in, __bf16* __restrict__ out, int K, int N) {
    __shared__ __bf16 tile[32][33];
    const int kb = blockIdx.y * 32;
    const int nb = blockIdx.x * 32;
    const int tx = threadIdx.x & 31;
    const int ty = threadIdx.x >> 5;   // 0..7
#pragma unroll
    for (int i = ty; i < 32; i += 8)
        tile[i][tx] = (__bf16)in[(size_t)(kb + i) * N + nb + tx];
    __syncthreads();
#pragma unroll
    for (int i = ty; i < 32; i += 8)
        out[(size_t)(nb + i) * K + kb + tx] = tile[tx][i];
}

// ---------------------------------------------------------------------------
// LayerNorm: one block per row, fp32 in, bf16 out
// ---------------------------------------------------------------------------
__global__ __launch_bounds__(256) void layernorm_kernel(const float* __restrict__ x,
                                                        const float* __restrict__ scale,
                                                        const float* __restrict__ shift,
                                                        __bf16* __restrict__ out) {
    const int row = blockIdx.x;
    const float* xr = x + (size_t)row * D_;
    __shared__ float red[256];
    const int tid = threadIdx.x;

    float s = 0.f;
    for (int i = tid; i < D_; i += 256) s += xr[i];
    red[tid] = s; __syncthreads();
    for (int off = 128; off > 0; off >>= 1) {
        if (tid < off) red[tid] += red[tid + off];
        __syncthreads();
    }
    const float mean = red[0] * (1.0f / D_);
    __syncthreads();

    float v = 0.f;
    for (int i = tid; i < D_; i += 256) { float d = xr[i] - mean; v += d * d; }
    red[tid] = v; __syncthreads();
    for (int off = 128; off > 0; off >>= 1) {
        if (tid < off) red[tid] += red[tid + off];
        __syncthreads();
    }
    const float inv = rsqrtf(red[0] * (1.0f / D_) + 1e-5f);

    for (int i = tid; i < D_; i += 256) {
        float r = scale[i] * (xr[i] - mean) * inv + shift[i];
        out[(size_t)row * D_ + i] = (__bf16)r;
    }
}

// ---------------------------------------------------------------------------
// Tiled bf16 WMMA GEMM: C[M,N] = A[M,K] @ Bt[N,K]^T  (Bt is pre-transposed)
// Block tile 64x128, BK=32, 256 threads = 8 waves (2x4), wave tile 32x32.
// Fused epilogue: +bias, exact GELU, +residual, fp32 and/or bf16 out.
// ---------------------------------------------------------------------------
__global__ __launch_bounds__(256) void gemm_bf16_kernel(const __bf16* __restrict__ A,
                                                        const __bf16* __restrict__ Bt,
                                                        float*  __restrict__ Cf,
                                                        __bf16* __restrict__ Cb,
                                                        const float* __restrict__ bias,
                                                        const float* __restrict__ residual,
                                                        int M, int N, int K, int do_gelu) {
    __shared__ __bf16 As[64][40];     // row stride 80B (16B multiple)
    __shared__ __bf16 BsT[128][40];   // rows = N, cols = K-chunk

    const int tid  = threadIdx.x;
    const int lane = tid & 31;
    const int wave = tid >> 5;
    const int wr   = wave >> 2;       // 0..1
    const int wc   = wave & 3;        // 0..3
    const int blockM = blockIdx.y * 64;
    const int blockN = blockIdx.x * 128;

    v8f acc[2][2];
#pragma unroll
    for (int i = 0; i < 2; ++i)
#pragma unroll
        for (int j = 0; j < 2; ++j) acc[i][j] = v8f_zero();

    for (int k0 = 0; k0 < K; k0 += 32) {
        // stage A tile 64x32 and Bt tile 128x32
        {
            const int ra = tid >> 2;
            const int ca = (tid & 3) * 8;
            const __bf16* sa = A + (size_t)(blockM + ra) * K + k0 + ca;
            const int rb = tid >> 1;
            const int cb = (tid & 1) * 16;
            const __bf16* sb = Bt + (size_t)(blockN + rb) * K + k0 + cb;
#ifdef HAVE_ASYNC_LDS
            async_copy_b128(sa, &As[ra][ca]);
            async_copy_b128(sb, &BsT[rb][cb]);
            async_copy_b128(sb + 8, &BsT[rb][cb + 8]);
#else
            *(uint4*)&As[ra][ca]       = *(const uint4*)sa;
            *(uint4*)&BsT[rb][cb]      = *(const uint4*)sb;
            *(uint4*)&BsT[rb][cb + 8]  = *(const uint4*)(sb + 8);
#endif
        }
        ASYNC_WAIT();
        __syncthreads();

        v16bf bfr[2];
#pragma unroll
        for (int nt = 0; nt < 2; ++nt)
            bfr[nt] = frag_row16(&BsT[wc * 32 + nt * 16 + (lane & 15)][0], lane);
#pragma unroll
        for (int mt = 0; mt < 2; ++mt) {
            const v16bf af = frag_row16(&As[wr * 32 + mt * 16 + (lane & 15)][0], lane);
#pragma unroll
            for (int nt = 0; nt < 2; ++nt)
                acc[mt][nt] = WMMA_BF16(af, bfr[nt], acc[mt][nt]);
        }
        __syncthreads();
    }

    // epilogue
    const int rowsel = (lane >> 4) * 8;
    const int col0   = lane & 15;
#pragma unroll
    for (int mt = 0; mt < 2; ++mt)
#pragma unroll
        for (int nt = 0; nt < 2; ++nt)
#pragma unroll
            for (int r = 0; r < 8; ++r) {
                const int row = blockM + wr * 32 + mt * 16 + rowsel + r;
                const int col = blockN + wc * 32 + nt * 16 + col0;
                float v = acc[mt][nt][r];
                if (bias)     v += bias[col];
                if (do_gelu)  v = 0.5f * v * (1.0f + erff(v * 0.70710678118f));
                if (residual) v += residual[(size_t)row * N + col];
                if (Cf) Cf[(size_t)row * N + col] = v;
                if (Cb) Cb[(size_t)row * N + col] = (__bf16)v;
            }
}

// ---------------------------------------------------------------------------
// Causal flash attention: grid = (T/64, B*H), 128 threads = 4 waves.
// Each wave owns 16 query rows. K tile row-major (contiguous B-frag for QK^T),
// V tile stored transposed (contiguous B-frag for PV).
// ---------------------------------------------------------------------------
__global__ __launch_bounds__(128) void attention_kernel(const __bf16* __restrict__ Q,
                                                        const __bf16* __restrict__ Kk,
                                                        const __bf16* __restrict__ V,
                                                        __bf16* __restrict__ O) {
    const int qtile = blockIdx.x;
    const int bh    = blockIdx.y;
    const int b     = bh / H_;
    const int h     = bh % H_;
    const int tid   = threadIdx.x;
    const int lane  = tid & 31;
    const int wave  = tid >> 5;
    const int qbase = qtile * 64;
    const size_t baseoff = (size_t)b * T_ * D_ + (size_t)h * HD_;

    __shared__ __bf16 Kt[64][72];    // [key][hd], stride 144B
    __shared__ __bf16 VtT[64][72];   // [hd][key], stride 144B
    __shared__ __bf16 Ps[4][16][72];

    const int rowsel = (lane >> 4) * 8;
    const int col0   = lane & 15;

    // Q fragments (16 rows x HD, two K-chunks of 32) straight from global
    v16bf qfrag[2];
    {
        const __bf16* qrow = Q + baseoff + (size_t)(qbase + wave * 16 + (lane & 15)) * D_;
#pragma unroll
        for (int kc = 0; kc < 2; ++kc) qfrag[kc] = frag_row16(qrow + kc * 32, lane);
    }

    v8f o_acc[4];
#pragma unroll
    for (int i = 0; i < 4; ++i) o_acc[i] = v8f_zero();
    float m_run[8], l_run[8];
#pragma unroll
    for (int r = 0; r < 8; ++r) { m_run[r] = -3.0e38f; l_run[r] = 0.f; }

    const float sc  = 0.125f;   // 1/sqrt(64)
    const int   nkt = qtile + 1;

    for (int kt = 0; kt < nkt; ++kt) {
        const int kstart = kt * 64;
        // cooperative staging: 2 threads per row, 32 bf16 each
        {
            const int r = tid >> 1;
            const int c = (tid & 1) * 32;
            const __bf16* kr = Kk + baseoff + (size_t)(kstart + r) * D_ + c;
            const __bf16* vr = V  + baseoff + (size_t)(kstart + r) * D_ + c;
#ifdef HAVE_ASYNC_LDS
#pragma unroll
            for (int j = 0; j < 4; ++j) async_copy_b128(kr + j * 8, &Kt[r][c + j * 8]);
#else
#pragma unroll
            for (int j = 0; j < 4; ++j)
                *(uint4*)&Kt[r][c + j * 8] = *(const uint4*)(kr + j * 8);
#endif
            // V: load rows, scatter transposed into VtT[hd][key]
#pragma unroll
            for (int j = 0; j < 4; ++j) {
                const v8bf vv = *(const v8bf*)(vr + j * 8);
#pragma unroll
                for (int e = 0; e < 8; ++e) VtT[c + j * 8 + e][r] = vv[e];
            }
        }
        ASYNC_WAIT();
        __syncthreads();

        // S = Q @ K^T : 4 column tiles of 16 keys
        v8f s_acc[4];
#pragma unroll
        for (int nt = 0; nt < 4; ++nt) {
            v8f c = v8f_zero();
#pragma unroll
            for (int kc = 0; kc < 2; ++kc) {
                // B[kk,n] = K[key=n][hd=kk] -> contiguous read of Kt row
                const v16bf bf = frag_row16(&Kt[nt * 16 + (lane & 15)][kc * 32], lane);
                c = WMMA_BF16(qfrag[kc], bf, c);
            }
            s_acc[nt] = c;
        }

        // scale + causal mask
#pragma unroll
        for (int nt = 0; nt < 4; ++nt)
#pragma unroll
            for (int r = 0; r < 8; ++r) {
                const int qrow = qbase + wave * 16 + rowsel + r;
                const int kcol = kstart + nt * 16 + col0;
                const float s  = s_acc[nt][r] * sc;
                s_acc[nt][r] = (kcol > qrow) ? -1.0e30f : s;
            }

        // row max across tiles, then across the 16 lanes of the half-wave
        float mx[8];
#pragma unroll
        for (int r = 0; r < 8; ++r) {
            float m = s_acc[0][r];
#pragma unroll
            for (int nt = 1; nt < 4; ++nt) m = fmaxf(m, s_acc[nt][r]);
            mx[r] = m;
        }
#pragma unroll
        for (int mask = 1; mask <= 8; mask <<= 1)
#pragma unroll
            for (int r = 0; r < 8; ++r) mx[r] = fmaxf(mx[r], __shfl_xor(mx[r], mask, 32));

        // online softmax update
        float alpha[8];
#pragma unroll
        for (int r = 0; r < 8; ++r) {
            const float mn = fmaxf(m_run[r], mx[r]);
            alpha[r] = expf(m_run[r] - mn);
            m_run[r] = mn;
        }
        float rs[8];
#pragma unroll
        for (int r = 0; r < 8; ++r) rs[r] = 0.f;
#pragma unroll
        for (int nt = 0; nt < 4; ++nt)
#pragma unroll
            for (int r = 0; r < 8; ++r) {
                const float p = expf(s_acc[nt][r] - m_run[r]);
                s_acc[nt][r] = p;
                rs[r] += p;
            }
#pragma unroll
        for (int mask = 1; mask <= 8; mask <<= 1)
#pragma unroll
            for (int r = 0; r < 8; ++r) rs[r] += __shfl_xor(rs[r], mask, 32);
#pragma unroll
        for (int r = 0; r < 8; ++r) l_run[r] = l_run[r] * alpha[r] + rs[r];
#pragma unroll
        for (int nt = 0; nt < 4; ++nt)
#pragma unroll
            for (int r = 0; r < 8; ++r) o_acc[nt][r] *= alpha[r];

        // P (C-layout) -> LDS -> A-layout
#pragma unroll
        for (int nt = 0; nt < 4; ++nt)
#pragma unroll
            for (int r = 0; r < 8; ++r)
                Ps[wave][rowsel + r][nt * 16 + col0] = (__bf16)s_acc[nt][r];
        __syncthreads();

        // O += P @ V
#pragma unroll
        for (int kc = 0; kc < 2; ++kc) {
            const v16bf af = frag_row16(&Ps[wave][lane & 15][kc * 32], lane);
#pragma unroll
            for (int nt = 0; nt < 4; ++nt) {
                // B[kk,n] = V[key=kk][hd=n] = VtT[n][kk] -> contiguous row read
                const v16bf bf = frag_row16(&VtT[nt * 16 + (lane & 15)][kc * 32], lane);
                o_acc[nt] = WMMA_BF16(af, bf, o_acc[nt]);
            }
        }
        __syncthreads();
    }

    // normalize + write out
#pragma unroll
    for (int r = 0; r < 8; ++r) l_run[r] = 1.0f / l_run[r];
#pragma unroll
    for (int nt = 0; nt < 4; ++nt)
#pragma unroll
        for (int r = 0; r < 8; ++r) {
            const int row = qbase + wave * 16 + rowsel + r;
            const int col = nt * 16 + col0;
            O[baseoff + (size_t)row * D_ + col] = (__bf16)(o_acc[nt][r] * l_run[r]);
        }
}

// ---------------------------------------------------------------------------
// Host-side orchestration
// ---------------------------------------------------------------------------
extern "C" void kernel_launch(void* const* d_in, const int* in_sizes, int n_in,
                              void* d_out, int out_size, void* d_ws, size_t ws_size,
                              hipStream_t stream) {
    const float* x    = (const float*)d_in[0];
    const float* Wq   = (const float*)d_in[1];
    const float* Wk   = (const float*)d_in[2];
    const float* Wv   = (const float*)d_in[3];
    const float* Wo   = (const float*)d_in[4];
    const float* bo   = (const float*)d_in[5];
    const float* W1   = (const float*)d_in[6];
    const float* b1   = (const float*)d_in[7];
    const float* W2   = (const float*)d_in[8];
    const float* b2   = (const float*)d_in[9];
    const float* ln1s = (const float*)d_in[10];
    const float* ln1b = (const float*)d_in[11];
    const float* ln2s = (const float*)d_in[12];
    const float* ln2b = (const float*)d_in[13];
    float* out = (float*)d_out;

    // carve workspace
    char* p = (char*)d_ws;
    auto alloc = [&](size_t bytes) -> void* {
        void* r = (void*)p;
        p += (bytes + 255) & ~(size_t)255;
        return r;
    };
    __bf16* Wq_t = (__bf16*)alloc((size_t)D_ * D_ * 2);   // transposed [N][K]
    __bf16* Wk_t = (__bf16*)alloc((size_t)D_ * D_ * 2);
    __bf16* Wv_t = (__bf16*)alloc((size_t)D_ * D_ * 2);
    __bf16* Wo_t = (__bf16*)alloc((size_t)D_ * D_ * 2);
    __bf16* W1_t = (__bf16*)alloc((size_t)D_ * D4_ * 2);
    __bf16* W2_t = (__bf16*)alloc((size_t)D4_ * D_ * 2);
    __bf16* h_b  = (__bf16*)alloc((size_t)M_ * D_ * 2);
    __bf16* q_b  = (__bf16*)alloc((size_t)M_ * D_ * 2);
    __bf16* k_b  = (__bf16*)alloc((size_t)M_ * D_ * 2);
    __bf16* v_b  = (__bf16*)alloc((size_t)M_ * D_ * 2);
    __bf16* ao_b = (__bf16*)alloc((size_t)M_ * D_ * 2);
    float*  x1   = (float*) alloc((size_t)M_ * D_ * 4);
    __bf16* h2_b = (__bf16*)alloc((size_t)M_ * D_ * 2);
    __bf16* ff_b = (__bf16*)alloc((size_t)M_ * D4_ * 2);

    // weight prep: convert + transpose ([K][N] fp32 -> [N][K] bf16)
    auto cvtT = [&](const float* src, __bf16* dst, int K, int N) {
        dim3 g(N / 32, K / 32);
        transpose_f32_to_bf16_kernel<<<g, 256, 0, stream>>>(src, dst, K, N);
    };
    cvtT(Wq, Wq_t, D_, D_);
    cvtT(Wk, Wk_t, D_, D_);
    cvtT(Wv, Wv_t, D_, D_);
    cvtT(Wo, Wo_t, D_, D_);
    cvtT(W1, W1_t, D_, D4_);
    cvtT(W2, W2_t, D4_, D_);

    // LN1
    layernorm_kernel<<<M_, 256, 0, stream>>>(x, ln1s, ln1b, h_b);

    // QKV projections (bf16 out, no bias)
    dim3 gDD(D_ / 128, M_ / 64);
    gemm_bf16_kernel<<<gDD, 256, 0, stream>>>(h_b, Wq_t, nullptr, q_b, nullptr, nullptr, M_, D_, D_, 0);
    gemm_bf16_kernel<<<gDD, 256, 0, stream>>>(h_b, Wk_t, nullptr, k_b, nullptr, nullptr, M_, D_, D_, 0);
    gemm_bf16_kernel<<<gDD, 256, 0, stream>>>(h_b, Wv_t, nullptr, v_b, nullptr, nullptr, M_, D_, D_, 0);

    // causal attention
    dim3 gAttn(T_ / 64, BN_ * H_);
    attention_kernel<<<gAttn, 128, 0, stream>>>(q_b, k_b, v_b, ao_b);

    // output projection: x1 = x + ao @ Wo + bo  (fp32 out)
    gemm_bf16_kernel<<<gDD, 256, 0, stream>>>(ao_b, Wo_t, x1, nullptr, bo, x, M_, D_, D_, 0);

    // LN2
    layernorm_kernel<<<M_, 256, 0, stream>>>(x1, ln2s, ln2b, h2_b);

    // FFN1: gelu(h2 @ W1 + b1) -> bf16
    dim3 gD4(D4_ / 128, M_ / 64);
    gemm_bf16_kernel<<<gD4, 256, 0, stream>>>(h2_b, W1_t, nullptr, ff_b, b1, nullptr, M_, D4_, D_, 1);

    // FFN2: out = x1 + ff @ W2 + b2  (fp32 -> d_out)
    gemm_bf16_kernel<<<gDD, 256, 0, stream>>>(ff_b, W2_t, out, nullptr, b2, x1, M_, D_, D4_, 0);

    (void)in_sizes; (void)n_in; (void)out_size; (void)ws_size;
}
